// ASD_39642548142294
// MI455X (gfx1250) — matrix operand order repaired and checked by reading, not compiled, not run on hardware
//
#include <hip/hip_runtime.h>
#include <hip/hip_bf16.h>

typedef __attribute__((ext_vector_type(16))) __bf16        v16bf;
typedef __attribute__((ext_vector_type(8)))  float         v8f;
typedef __attribute__((ext_vector_type(8)))  unsigned int  v8u;
typedef __attribute__((ext_vector_type(4)))  unsigned int  v4u;
typedef __attribute__((ext_vector_type(8)))  int           v8i;
typedef __attribute__((ext_vector_type(4)))  int           v4i;

#define FEAT 2048
#define HID  1024
#define G4H  4096
#define BATCH 32
#define T    512
#define OUTD 2
#define NBLK 32           // recurrent workgroups (grid-sync domain)
#define HS   32           // hidden units per recurrent WG
#define WHH_STRIDE 1032   // bf16 elems per LDS row: 1024 + 8 pad (bank decorrelation)
#define GS_ELEMS (128 * 32)   // per-step gate slice: 128 local gates x 32 batch (bf16)

#define SMEM_BYTES (128 * WHH_STRIDE * 2 /*whh*/ + 128 * 33 * 4 /*gbuf*/ + \
                    32 * 33 * 4 /*hbuf*/ + 64 * 4 /*wfc*/ + 2 * GS_ELEMS * 2 /*gslice db*/)

#define WMMA_BF16(a, b, c) \
  __builtin_amdgcn_wmma_f32_16x16x32_bf16(false, (a), false, (b), (short)0, (c), false, false)

#if defined(__has_builtin)
#if __has_builtin(__builtin_amdgcn_tensor_load_to_lds) && \
    __has_builtin(__builtin_amdgcn_s_wait_tensorcnt)
#define USE_TDM 1
#endif
#endif
#ifndef USE_TDM
#define USE_TDM 0
#endif

// ---- fragment builders (contiguous k = 16*half + s in both A and B, wave32) ----

// A (16x32, M x K), f32 source row-major, converting to bf16.
__device__ __forceinline__ v16bf frag_a_f32(const float* base, int ldk, int k0) {
  const int lane = threadIdx.x & 31;
  const int hl = lane >> 4;
  const float* row = base + (size_t)(lane & 15) * ldk + k0;
  v16bf a;
#pragma unroll
  for (int j = 0; j < 8; ++j) {
    const int k = (hl << 4) + (j << 1);
    float2 v = *(const float2*)(row + k);
    a[2 * j]     = (__bf16)v.x;
    a[2 * j + 1] = (__bf16)v.y;
  }
  return a;
}

// A (16x32) from bf16 source (h state); 2 bf16 per u32, merges to b128 loads.
__device__ __forceinline__ v16bf frag_a_bf16(const __bf16* base, int ldk, int k0) {
  const int lane = threadIdx.x & 31;
  const int hl = lane >> 4;
  const __bf16* row = base + (size_t)(lane & 15) * ldk + k0;
  v8u raw;
#pragma unroll
  for (int j = 0; j < 8; ++j) {
    const int k = (hl << 4) + (j << 1);
    raw[j] = *(const unsigned int*)(row + k);
  }
  return __builtin_bit_cast(v16bf, raw);
}

// B (32x16, K x N) built from W stored [N][K] row-major (i.e. B = W^T).
__device__ __forceinline__ v16bf frag_b_f32(const float* base, int ldk, int k0) {
  const int lane = threadIdx.x & 31;
  const int hl = lane >> 4;
  const float* row = base + (size_t)(lane & 15) * ldk + k0;
  v16bf b;
#pragma unroll
  for (int p = 0; p < 8; ++p) {
    const int k = (hl << 4) + (p << 1);
    float2 v = *(const float2*)(row + k);
    b[2 * p]     = (__bf16)v.x;
    b[2 * p + 1] = (__bf16)v.y;
  }
  return b;
}

// B (32x16) from an LDS-resident bf16 row (one gate row of W_hh).
__device__ __forceinline__ v16bf frag_b_lds(const __bf16* row, int k0) {
  const int lane = threadIdx.x & 31;
  const int hl = lane >> 4;
  v8u raw;
#pragma unroll
  for (int p = 0; p < 8; ++p) {
    raw[p] = *(const unsigned int*)(row + k0 + (hl << 4) + (p << 1));
  }
  return __builtin_bit_cast(v16bf, raw);
}

#if USE_TDM
// raw LDS byte offset (D# wants it), via generic -> addrspace(3) cast
__device__ __forceinline__ unsigned lds_offset_of(const void* p) {
  return (unsigned)(size_t)(__attribute__((address_space(3))) const char*)p;
}

// TDM: 3-D tile  dim0 = 32 gates (contig) | dim1 = 4 gate groups @ stride 1024
//                dim2 = 32 batch rows @ stride 512*4096   (all bf16 elements)
__device__ __forceinline__ void tdm_load_gslice(const __bf16* gptr, unsigned lds_off) {
  const unsigned long long ga = (unsigned long long)(size_t)gptr;
  v4u g0 = {1u,                                   // count=1, user descriptor
            lds_off,                              // lds_addr
            (unsigned)ga,                         // global_addr[31:0]
            (unsigned)((ga >> 32) & 0x01FFFFFFu) | (2u << 30)};  // addr[56:32] | type=2
  v8i g1 = {(int)(1u << 16),        // workgroup_mask=0 | data_size=1 (2B)
            (int)(32u << 16),       // atomic_barrier_addr=0 | tensor_dim0.lo16 = 32
            (int)(4u << 16),        // tensor_dim0.hi16=0 | tensor_dim1.lo16 = 4
            (int)(32u << 16),       // tensor_dim1.hi16=0 | tile_dim0 = 32
            (int)(4u | (32u << 16)),// tile_dim1 = 4 | tile_dim2 = 32
            1024,                   // tensor_dim0_stride.lo32 = 1024
            0,                      // stride0.hi16=0 | tensor_dim1_stride.lo16=0
            0x20};                  // tensor_dim1_stride[47:16] = 0x20 (=512*4096)
  v4i g2 = {32, 0, 0, 0};           // tensor_dim2 = 32; dim3/tile_dim3 unused
  v4i g3 = {0, 0, 0, 0};
#if __clang_major__ >= 23
  v8i g4 = {0, 0, 0, 0, 0, 0, 0, 0};
  __builtin_amdgcn_tensor_load_to_lds(g0, g1, g2, g3, g4, 0);
#else
  __builtin_amdgcn_tensor_load_to_lds(g0, g1, g2, g3, 0);
#endif
}
#endif

// ---- kernel 0: initialize out (=b_fc), h0 (=0), sync counters (=0) ----
__global__ void asd_init(float* __restrict__ out, const float* __restrict__ bfc,
                         unsigned int* __restrict__ syncnt, unsigned int* __restrict__ Hz) {
  const int i = blockIdx.x * blockDim.x + threadIdx.x;
  if (i < BATCH * T * OUTD) out[i] = bfc[i & 1];
  if (i < T) syncnt[i] = 0u;
  if (i < (2 * BATCH * HID) / 2) Hz[i] = 0u;  // both h ping-pong buffers (bf16 pairs)
}

// ---- kernel 1: G[r][g] = X @ W_ih^T + (b_ih + b_hh), r = b*512 + t, stored bf16 ----
__global__ __launch_bounds__(256) void asd_gemm_xgates(
    const float* __restrict__ X, const float* __restrict__ Wih,
    const float* __restrict__ bih, const float* __restrict__ bhh,
    __bf16* __restrict__ G) {
  const int wave = threadIdx.x >> 5;
  const int lane = threadIdx.x & 31;
  const int hl = lane >> 4, nn = lane & 15;
  const int M0 = blockIdx.x * 64 + (wave & 1) * 32;   // r (batch*seq rows)
  const int N0 = blockIdx.y * 128 + (wave >> 1) * 32; // gate cols

  v8f acc[2][2] = {};
  for (int k0 = 0; k0 < FEAT; k0 += 32) {
    v16bf a0 = frag_a_f32(X + (size_t)M0 * FEAT, FEAT, k0);
    v16bf a1 = frag_a_f32(X + (size_t)(M0 + 16) * FEAT, FEAT, k0);
    v16bf b0 = frag_b_f32(Wih + (size_t)N0 * FEAT, FEAT, k0);
    v16bf b1 = frag_b_f32(Wih + (size_t)(N0 + 16) * FEAT, FEAT, k0);
    acc[0][0] = WMMA_BF16(a0, b0, acc[0][0]);
    acc[0][1] = WMMA_BF16(a0, b1, acc[0][1]);
    acc[1][0] = WMMA_BF16(a1, b0, acc[1][0]);
    acc[1][1] = WMMA_BF16(a1, b1, acc[1][1]);
  }
#pragma unroll
  for (int nt = 0; nt < 2; ++nt) {
    const int g = N0 + nt * 16 + nn;
    const float bias = bih[g] + bhh[g];
#pragma unroll
    for (int mt = 0; mt < 2; ++mt) {
#pragma unroll
      for (int i = 0; i < 8; ++i) {
        const int r = M0 + mt * 16 + i + 8 * hl;  // C layout: M = i + 8*half, N = lane&15
        G[(size_t)r * G4H + g] = (__bf16)(acc[mt][nt][i] + bias);
      }
    }
  }
}

// ---- kernel 2: persistent recurrent LSTM; 32 WGs, W_hh slice resident in LDS ----
__global__ __launch_bounds__(256) void asd_lstm(
    const __bf16* __restrict__ G, const float* __restrict__ Whh,
    const float* __restrict__ Wfc, float* __restrict__ out,
    __bf16* __restrict__ H, unsigned int* __restrict__ syncnt) {
  extern __shared__ char smem[];
  __bf16* whh = (__bf16*)smem;                               // [128][WHH_STRIDE] bf16
  float*  gbuf = (float*)(smem + 128 * WHH_STRIDE * 2);      // [128][33] gate exchange
  float*  hbuf = gbuf + 128 * 33;                            // [32][33] h slice
  float*  wfc  = hbuf + 32 * 33;                             // [2][32]
  __bf16* gslice = (__bf16*)(wfc + 64);                      // [2][32 b][128 lg] db

  const int tid = threadIdx.x;
  const int hs0 = blockIdx.x * HS;
  const int wave = tid >> 5, lane = tid & 31;
  const int hl = lane >> 4, nn = lane & 15;

  // Preload this WG's 128 gate rows of W_hh (rows q*1024 + hs0 + jj) as bf16 into LDS.
  for (int idx = tid; idx < 128 * HID; idx += 256) {
    const int l = idx >> 10;
    const int k = idx & (HID - 1);
    const int gr = ((l >> 5) << 10) + hs0 + (l & 31);
    whh[l * WHH_STRIDE + k] = (__bf16)Whh[(size_t)gr * HID + k];
  }
  if (tid < 64) wfc[tid] = Wfc[(size_t)(tid >> 5) * HID + hs0 + (tid & 31)];

  const int ln = wave * 16 + nn;                 // local gate col 0..127 (wave = N tile)
  const int grow = ((ln >> 5) << 10) + hs0 + (ln & 31);  // global gate row
  const __bf16* brow = whh + ln * WHH_STRIDE;

  const int ej = tid >> 3;                       // elementwise: hidden idx 0..31
  const int eb0 = tid & 7;                       // batch group
  float creg[4] = {0.f, 0.f, 0.f, 0.f};          // cell state in registers all 512 steps

#if USE_TDM
  const unsigned gs_off = lds_offset_of(gslice);
  if (wave == 0) tdm_load_gslice(G + hs0, gs_off);  // t = 0 slice into buffer 0
#endif
  __syncthreads();

  for (int t = 0; t < T; ++t) {
    const __bf16* Hc = H + (size_t)(t & 1) * (BATCH * HID);
    __bf16* Hn = (__bf16*)(H + (size_t)((t + 1) & 1) * (BATCH * HID));

#if USE_TDM
    // DMA next step's gate slice while this step computes; wait for current slice.
    if (wave == 0) {
      if (t + 1 < T) {
        tdm_load_gslice(G + (size_t)(t + 1) * G4H + hs0,
                        gs_off + (unsigned)(((t + 1) & 1) * GS_ELEMS * 2));
        __builtin_amdgcn_s_wait_tensorcnt(1);
      } else {
        __builtin_amdgcn_s_wait_tensorcnt(0);
      }
    }
    __syncthreads();  // publish TDM-written LDS slice to all waves
    const __bf16* gs = gslice + (t & 1) * GS_ELEMS;  // [b][q][jj] => [b*128 + ln]
#else
    // software-prefetch next step's slice (128 distinct lines per WG)
    if (t + 1 < T && tid < 128) {
      const int pb = tid & 31, pq = tid >> 5;
      __builtin_prefetch(&G[(size_t)(pb * T + t + 1) * G4H + (pq << 10) + hs0], 0, 3);
    }
#endif

    // init accumulators with precomputed x-gates (+bias already folded in)
    v8f acc[2];
#pragma unroll
    for (int mt = 0; mt < 2; ++mt) {
#pragma unroll
      for (int i = 0; i < 8; ++i) {
        const int b = mt * 16 + i + 8 * hl;
#if USE_TDM
        acc[mt][i] = (float)gs[b * 128 + ln];
#else
        acc[mt][i] = (float)G[(size_t)(b * T + t) * G4H + grow];
#endif
      }
    }

    // h @ W_hh^T for this gate slice: K = 1024, B fragments from LDS-resident W_hh
    for (int k0 = 0; k0 < HID; k0 += 32) {
      v16bf bf = frag_b_lds(brow, k0);
      v16bf a0 = frag_a_bf16(Hc, HID, k0);
      v16bf a1 = frag_a_bf16(Hc + 16 * HID, HID, k0);
      acc[0] = WMMA_BF16(a0, bf, acc[0]);
      acc[1] = WMMA_BF16(a1, bf, acc[1]);
    }

#pragma unroll
    for (int mt = 0; mt < 2; ++mt) {
#pragma unroll
      for (int i = 0; i < 8; ++i) {
        const int b = mt * 16 + i + 8 * hl;
        gbuf[ln * 33 + b] = acc[mt][i];
      }
    }
    __syncthreads();

    // LSTM cell for hidden hs0+ej, 4 batch elements per thread
#pragma unroll
    for (int u = 0; u < 4; ++u) {
      const int b = eb0 + (u << 3);
      const float ig = gbuf[(0 * 32 + ej) * 33 + b];
      const float fg = gbuf[(1 * 32 + ej) * 33 + b];
      const float gg = gbuf[(2 * 32 + ej) * 33 + b];
      const float og = gbuf[(3 * 32 + ej) * 33 + b];
      const float is = 1.f / (1.f + __expf(-ig));
      const float fs = 1.f / (1.f + __expf(-fg));
      const float gt = tanhf(gg);
      const float os = 1.f / (1.f + __expf(-og));
      const float c = fs * creg[u] + is * gt;
      creg[u] = c;
      const float h = os * tanhf(c);
      Hn[(size_t)b * HID + hs0 + ej] = (__bf16)h;
      hbuf[ej * 33 + b] = h;
    }
    __threadfence();   // make this thread's h stores device-visible before the barrier
    __syncthreads();

    // fused FC readout: partial dot over this WG's 32 hidden units
    if (tid < 64) {
      const int b = tid & 31, o = tid >> 5;
      float s = 0.f;
#pragma unroll
      for (int j = 0; j < 32; ++j) s += hbuf[j * 33 + b] * wfc[o * 32 + j];
      atomicAdd(&out[(size_t)(b * T + t) * OUTD + o], s);
    }

    // grid-wide barrier across the 32 persistent WGs (fresh counter per step)
    if (tid == 0) {
      __hip_atomic_fetch_add(&syncnt[t], 1u, __ATOMIC_RELEASE, __HIP_MEMORY_SCOPE_AGENT);
      while (__hip_atomic_load(&syncnt[t], __ATOMIC_ACQUIRE, __HIP_MEMORY_SCOPE_AGENT) <
             (unsigned)NBLK) {
        __builtin_amdgcn_s_sleep(2);
      }
    }
    __syncthreads();
    __threadfence();   // acquire side: don't read stale h from near caches
  }
}

extern "C" void kernel_launch(void* const* d_in, const int* in_sizes, int n_in,
                              void* d_out, int out_size, void* d_ws, size_t ws_size,
                              hipStream_t stream) {
  const float* x   = (const float*)d_in[0];
  const float* Wih = (const float*)d_in[1];
  const float* Whh = (const float*)d_in[2];
  const float* bih = (const float*)d_in[3];
  const float* bhh = (const float*)d_in[4];
  const float* Wfc = (const float*)d_in[5];
  const float* bfc = (const float*)d_in[6];
  float* out = (float*)d_out;

  // workspace: G (bf16, 128 MB) | H ping-pong (bf16, 128 KB) | sync counters (2 KB)
  __bf16* G = (__bf16*)d_ws;
  __bf16* H = G + (size_t)BATCH * T * G4H;
  unsigned int* syncnt = (unsigned int*)(H + 2 * BATCH * HID);

  asd_init<<<128, 256, 0, stream>>>(out, bfc, syncnt, (unsigned int*)H);
  asd_gemm_xgates<<<dim3((BATCH * T) / 64, G4H / 128), 256, 0, stream>>>(x, Wih, bih, bhh, G);
  asd_lstm<<<NBLK, 256, SMEM_BYTES, stream>>>(G, Whh, Wfc, out, H, syncnt);
}
